// FeatureAlign_51951924412991
// MI455X (gfx1250) — compile-verified, matrix operand order checked by reading, not run on hardware
//
#include <hip/hip_runtime.h>

// CDNA5 (gfx1250) fused deformable-conv implicit GEMM, round 3.
//   out[b,o,p] = relu( sum_{g,c,k} bilinear(x)[b,g,c,k,p] * w_deform[o,g,c,k] )
// - Kernel 1: prepack w_deform (fp32) -> bf16 tiles in d_ws, laid out per WMMA A-fragment.
// - Kernel 2: per block: 256(o) x 32(pixel) output tile; 72 K-steps of 32.
//   A-tile (16KB, contiguous) streamed to LDS via Tensor Data Mover (tensor_load_to_lds,
//   TENSORcnt), overlapped with the bilinear-gather B-tile build; 32 WMMAs/step.

typedef __attribute__((ext_vector_type(16))) __bf16 v16bf;
typedef __attribute__((ext_vector_type(8)))  __bf16 v8bf;
typedef __attribute__((ext_vector_type(8)))  float  v8f;
typedef __attribute__((ext_vector_type(4)))  unsigned int v4u;
typedef __attribute__((ext_vector_type(8)))  int    v8i;
typedef __attribute__((ext_vector_type(4)))  int    v4i;

#define Hdim 64
#define Wdim 64
#define HWp  (Hdim*Wdim)
#define Cin  256
#define Cout 256
#define DGn  4
#define Cg   64
#define KK   9
#define NT   32        // pixels per block (N tile)
#define MT   256       // output channels per block (M tile) = all of Cout
#define TILE_ELEMS (MT*32)      // bf16 elems per packed A tile (16 KB)

#if __has_builtin(__builtin_amdgcn_tensor_load_to_lds)
#define USE_TDM 1
#endif

// ---- prepack: wdef[o][ch][k] fp32 -> wpack[((g*9+k)*2+cb)][o][cl] bf16 ----
__global__ __launch_bounds__(256)
void pack_weights(const float* __restrict__ wdef, __bf16* __restrict__ wpack)
{
    int e = blockIdx.x * 256 + threadIdx.x;     // 72 tiles * 8192 = 589824 total
    int tile = e >> 13;
    int r    = e & 8191;
    int o    = r >> 5;
    int cl   = r & 31;
    int cb   = tile & 1;
    int gk   = tile >> 1;
    int g    = gk / KK;
    int k    = gk - g * KK;
    float wv = wdef[((size_t)o * Cin + (g * Cg + cb * 32 + cl)) * KK + k];
    wpack[e] = (__bf16)wv;
}

__global__ __launch_bounds__(256)
void deform_conv_wmma(const float* __restrict__ x,
                      const float* __restrict__ shp,
                      const float* __restrict__ woff,
                      const __bf16* __restrict__ wpack,
                      float* __restrict__ out)
{
    __shared__ __align__(16) float  s_shape[4][NT];
    __shared__ __align__(16) float  s_woff[DGn*KK*2*4];   // 288
    __shared__ __align__(16) int    s_sidx[KK][NT][4];    // corner indices
    __shared__ __align__(16) float  s_sw[KK][NT][4];      // bilinear weights (validity folded)
    __shared__ __align__(16) __bf16 s_A[MT*32];           // A tile [m][k], 64B rows (16 KB)
    __shared__ __align__(16) __bf16 s_B[NT*32];           // B tile transposed [n][k]

    const int tid  = threadIdx.x;
    const int lane = tid & 31;
    const int wave = tid >> 5;
    const int wrow = wave >> 1;            // 4 wave-rows, each covers 64 M rows (4 subtiles)
    const int wcol = wave & 1;             // 2 wave-cols cover N=32
    const int b      = blockIdx.z;
    const int p_base = blockIdx.x * NT;

    s_woff[tid] = woff[tid];
    if (tid < 32) s_woff[256 + tid] = woff[256 + tid];
    if (tid < 4 * NT) {
        int c = tid >> 5, p = tid & 31;
        s_shape[c][p] = shp[((size_t)b * 4 + c) * HWp + p_base + p];
    }

    v8f acc[4];                            // 4 16x16 f32 tiles per wave (32 VGPRs)
    #pragma unroll
    for (int t = 0; t < 4; ++t) acc[t] = (v8f){};

    __syncthreads();

    for (int g = 0; g < DGn; ++g) {
        // ---- sampling metadata: 9 taps x 32 pixels, offsets computed inline ----
        for (int item = tid; item < KK * NT; item += 256) {
            int k = item >> 5;
            int p = item & 31;
            int pg = p_base + p;
            float hh = (float)(pg >> 6);
            float ww = (float)(pg & 63);
            const float* wy = &s_woff[((g * KK + k) * 2 + 0) * 4];
            const float* wx = &s_woff[((g * KK + k) * 2 + 1) * 4];
            float oy = wy[0]*s_shape[0][p] + wy[1]*s_shape[1][p]
                     + wy[2]*s_shape[2][p] + wy[3]*s_shape[3][p];
            float ox = wx[0]*s_shape[0][p] + wx[1]*s_shape[1][p]
                     + wx[2]*s_shape[2][p] + wx[3]*s_shape[3][p];
            float py = oy + hh + (float)(k / 3 - 1);
            float px = ox + ww + (float)(k % 3 - 1);
            float y0f = floorf(py), x0f = floorf(px);
            float fy = py - y0f, fx = px - x0f;
            int y0 = (int)y0f, x0 = (int)x0f;
            float wcy[2] = {1.f - fy, fy};
            float wcx[2] = {1.f - fx, fx};
            #pragma unroll
            for (int dy = 0; dy < 2; ++dy) {
                #pragma unroll
                for (int dx = 0; dx < 2; ++dx) {
                    int yi = y0 + dy, xi = x0 + dx;
                    bool valid = (yi >= 0) && (yi < Hdim) && (xi >= 0) && (xi < Wdim);
                    int yc = min(max(yi, 0), Hdim - 1);
                    int xc = min(max(xi, 0), Wdim - 1);
                    s_sidx[k][p][dy * 2 + dx] = yc * Wdim + xc;
                    s_sw  [k][p][dy * 2 + dx] = valid ? (wcy[dy] * wcx[dx]) : 0.f;
                }
            }
        }
        __syncthreads();

        for (int k = 0; k < KK; ++k) {
            #pragma unroll
            for (int cb = 0; cb < 2; ++cb) {
                const int tile = ((g * KK + k) * 2 + cb);
                const __bf16* wp_tile = wpack + (size_t)tile * TILE_ELEMS;

#ifdef USE_TDM
                // ---- issue TDM: contiguous 16KB packed A tile -> LDS (async) ----
                if (wave == 0) {
                    unsigned long long ga = (unsigned long long)(uintptr_t)wp_tile;
                    unsigned la = (unsigned)(uintptr_t)(void*)&s_A[0];  // low32 = LDS offset
                    // D# group0: count=1 | lds_addr | global_addr[56:0] | type=2
                    v4u g0 = { 1u, la, (unsigned)ga,
                               (unsigned)((ga >> 32) & 0x1FFFFFFull) | 0x80000000u };
                    // D# group1: data_size=8B; tensor_dim0=2048; tensor_dim1=1;
                    //            tile_dim0=2048 (1-D tile); dim0_stride=2048
                    v8i g1 = { 0x00030000,
                               (int)0x08000000,   // tensor_dim0[15:0] << 16
                               (int)0x00010000,   // tensor_dim0[31:16]=0 | tensor_dim1[15:0]<<16
                               (int)0x08000000,   // tensor_dim1[31:16]=0 | tile_dim0<<16
                               0,                 // tile_dim1=0, tile_dim2=0
                               2048,              // tensor_dim0_stride[31:0]
                               0, 0 };
                    v4i gz4 = { 0, 0, 0, 0 };
                    v8i gz8 = { 0, 0, 0, 0, 0, 0, 0, 0 };
                    __builtin_amdgcn_tensor_load_to_lds(g0, g1, gz4, gz4, gz8, 0);
                }
#endif
                // ---- build B tile (overlaps the TDM): bilinear gathers ----
                #pragma unroll
                for (int j = 0; j < 4; ++j) {
                    int e = tid + 256 * j;       // 1024 elements
                    int p = e & 31, c = e >> 5;
                    const float* xc = x + (((size_t)b * Cin + g * Cg + cb * 32 + c) * HWp);
                    const int*   id = s_sidx[k][p];
                    const float* sw = s_sw[k][p];
                    float v = sw[0] * xc[id[0]] + sw[1] * xc[id[1]]
                            + sw[2] * xc[id[2]] + sw[3] * xc[id[3]];
                    s_B[p * 32 + c] = (__bf16)v;
                }
#ifdef USE_TDM
                if (wave == 0) __builtin_amdgcn_s_wait_tensorcnt(0);
#else
                // fallback: cooperative 16B copies of the packed tile
                #pragma unroll
                for (int j = 0; j < 4; ++j) {
                    int e = tid + 256 * j;       // 1024 x 16B = 16KB
                    *(uint4*)&s_A[e * 8] = ((const uint4*)wp_tile)[e];
                }
#endif
                __syncthreads();

                // ---- 4 WMMAs per wave; B fragment reused across M subtiles ----
                int kh = lane >> 4;
                int rB = wcol * 16 + (lane & 15);
                v8bf blo = *(const v8bf*)&s_B[rB * 32 + kh * 8];
                v8bf bhi = *(const v8bf*)&s_B[rB * 32 + 16 + kh * 8];
                v16bf bm = __builtin_shufflevector(blo, bhi,
                           0,1,2,3,4,5,6,7,8,9,10,11,12,13,14,15);
                #pragma unroll
                for (int t = 0; t < 4; ++t) {
                    int rA = wrow * 64 + t * 16 + (lane & 15);
                    v8bf alo = *(const v8bf*)&s_A[rA * 32 + kh * 8];
                    v8bf ahi = *(const v8bf*)&s_A[rA * 32 + 16 + kh * 8];
                    v16bf a = __builtin_shufflevector(alo, ahi,
                              0,1,2,3,4,5,6,7,8,9,10,11,12,13,14,15);
                    acc[t] = __builtin_amdgcn_wmma_f32_16x16x32_bf16(
                                 false, a, false, bm, (short)0, acc[t], false, false);
                }
                __syncthreads();
            }
        }
    }

    // ---- ReLU + store per C/D layout (lane<16: M=i, lane>=16: M=8+i; N=lane&15) ----
    {
        int p = p_base + wcol * 16 + (lane & 15);
        #pragma unroll
        for (int t = 0; t < 4; ++t) {
            int o0 = wrow * 64 + t * 16 + ((lane >> 4) * 8);
            #pragma unroll
            for (int i = 0; i < 8; ++i) {
                float v = acc[t][i];
                v = v > 0.f ? v : 0.f;
                out[((size_t)b * Cout + o0 + i) * HWp + p] = v;
            }
        }
    }
}

extern "C" void kernel_launch(void* const* d_in, const int* in_sizes, int n_in,
                              void* d_out, int out_size, void* d_ws, size_t ws_size,
                              hipStream_t stream) {
    const float* x    = (const float*)d_in[0];   // [8,256,64,64]
    const float* shp  = (const float*)d_in[1];   // [8,4,64,64]
    const float* woff = (const float*)d_in[2];   // [72,4,1,1]
    const float* wdef = (const float*)d_in[3];   // [256,256,3,3]
    float* out   = (float*)d_out;                // [8,256,64,64]
    __bf16* wpack = (__bf16*)d_ws;               // 1.18 MB packed bf16 weights

    pack_weights<<<dim3(2304), dim3(256), 0, stream>>>(wdef, wpack);

    dim3 grid(HWp / NT, 1, 8);                   // (128, 1, 8) = 1024 blocks
    deform_conv_wmma<<<grid, dim3(256), 0, stream>>>(x, shp, woff, wpack, out);
}
